// HybridAblationModel_23433341567739
// MI455X (gfx1250) — matrix operand order repaired and checked by reading, not compiled, not run on hardware
//
#include <hip/hip_runtime.h>
#include <hip/hip_bf16.h>

typedef __bf16 v16bf __attribute__((ext_vector_type(16)));
typedef float  v8f   __attribute__((ext_vector_type(8)));

#define HDIM  1024
#define SROW  1028          // padded LDS row stride (floats): kills bank conflicts
#define DD    6
#define MT    32            // rows per workgroup (2 WMMA m-tiles)
#define NROWS 40960         // (32*256)*5
#define ASTR  10            // u32 per lane in A-staging (8 used + 2 pad -> conflict-free)
#define AKC   (32 * ASTR)   // u32 per (k-chunk, m-tile) block

// Cayley sign for e_a * e_b -> e_{a^b} in Cl(4,1): reorder sign + metric (-1 only on bit 4)
__device__ __forceinline__ float cay_sign(int a, int b) {
    int s = 0;
    int t = a >> 1;
    while (t) { s += __popc(t & b); t >>= 1; }
    float sg = (s & 1) ? -1.f : 1.f;
    if ((a & b) & 16) sg = -sg;          // METRIC = diag(1,1,1,1,-1)
    return sg;
}

// q[c] = reverse-sign(c) * CAYLEY[c,c,0]; constant-folds for literal c
__device__ __forceinline__ float blade_q(int c) {
    int k = __popc(c);
    float rev = ((k * (k - 1) / 2) & 1) ? -1.f : 1.f;
    return rev * cay_sign(c, c);
}

// -----------------------------------------------------------------------------
// Kernel 1: Kmat[i*32+b][o*32+c] = sign(b^c,b) * w[i,o,b^c], stored bf16 in the
// WMMA B-fragment layout: block (kc,nt) = 32x16 tile; lane = 16*(K>=16)+N,
// 16 contiguous bf16 per lane covering K = kc*32 + (lane>=16?16:0) + e.
// -----------------------------------------------------------------------------
__global__ __launch_bounds__(256) void build_k_swz(const float* __restrict__ gaw,
                                                   __bf16* __restrict__ kswz) {
    int flat = blockIdx.x * 256 + threadIdx.x;      // 0 .. 2^20-1
    int e    = flat & 15;
    int lane = (flat >> 4) & 31;
    int blk  = flat >> 9;
    int nt   = blk & 63;
    int kc   = blk >> 6;
    int row  = kc * 32 + ((lane >> 4) << 4) + e;    // = i*32 + b
    int col  = nt * 16 + (lane & 15);               // = o*32 + c
    int i = row >> 5, bb = row & 31;
    int o = col >> 5, cc = col & 31;
    int a = bb ^ cc;
    float v = cay_sign(a, bb) * gaw[((i << 5) + o) * 32 + a];
    kswz[flat] = (__bf16)v;
}

// per-row partial of  sum_c row[c] * w_out[c,:]  striped over 32 lanes
__device__ __forceinline__ void rowdot6(const float* __restrict__ row,
                                        const float* __restrict__ w_out,
                                        int lane, float acc[DD]) {
    for (int c = lane; c < HDIM; c += 32) {
        float h = row[c];
        const float* wr = w_out + c * DD;
        #pragma unroll
        for (int d = 0; d < DD; ++d) acc[d] += h * wr[d];
    }
}

// manifold norm of the 32x1024 tile in sH; optionally emit bf16 A-frag staging
template <bool STAGE>
__device__ __forceinline__ void norm_tile(float* sH, unsigned int* sAB, int tid) {
    const int r = tid & 31;
    #pragma unroll
    for (int it = 0; it < 4; ++it) {
        int kc = (tid >> 5) + (it << 3);            // channel == K-chunk index
        float* p = sH + r * SROW + (kc << 5);
        float v[32];
        float q = 0.f;
        #pragma unroll
        for (int c = 0; c < 32; ++c) { v[c] = p[c]; q += blade_q(c) * v[c] * v[c]; }
        float s = rsqrtf(fabsf(q) + 1e-6f);
        #pragma unroll
        for (int c = 0; c < 32; ++c) { v[c] *= s; p[c] = v[c]; }
        if (STAGE) {
            unsigned int* ab = sAB + (kc * 2 + (r >> 4)) * AKC;   // (kc, m-tile)
            int rr = r & 15;
            #pragma unroll
            for (int j = 0; j < 16; ++j) {
                int kloc = j << 1;
                int hb   = kloc >> 4;                               // lane-half bit
                int w    = (kloc & 15) >> 1;                        // dword slot
                union { __bf16 h[2]; unsigned int u; } pk;
                pk.h[0] = (__bf16)v[kloc];
                pk.h[1] = (__bf16)v[kloc + 1];
                ab[(hb * 16 + rr) * ASTR + w] = pk.u;
            }
        }
    }
}

// -----------------------------------------------------------------------------
// Kernel 2: fully fused per 32-row tile.
// -----------------------------------------------------------------------------
__global__ __launch_bounds__(256) void fused_ga(
        const float* __restrict__ x,     const float* __restrict__ w_in,
        const float* __restrict__ b_in,  const float* __restrict__ w_out,
        const float* __restrict__ b_out, const __bf16* __restrict__ kswz,
        float* __restrict__ out) {
    __shared__ float        sH[MT * SROW];        // 128.5 KB: hn, later reused for y
    __shared__ unsigned int sAB[32 * 2 * AKC];    // 80 KB: bf16 A-frags, WMMA layout
    __shared__ float        sX[MT * DD];

    const int tid  = threadIdx.x;
    const int lane = tid & 31;
    const int wv   = tid >> 5;                    // 8 waves
    const int rowBase = blockIdx.x * MT;

    if (tid < MT * DD) sX[tid] = x[(size_t)rowBase * DD + tid];
    __syncthreads();

    // ---- phase 1: h = x @ w_in + b_in  (K = 6)
    for (int idx = tid; idx < MT * HDIM; idx += 256) {
        int r = idx >> 10, c = idx & (HDIM - 1);
        float acc = b_in[c];
        #pragma unroll
        for (int d = 0; d < DD; ++d) acc += sX[r * DD + d] * w_in[d * HDIM + c];
        sH[r * SROW + c] = acc;
    }
    __syncthreads();

    // ---- phase 2: manifold norm in place + cooperative bf16 A-frag staging
    norm_tile<true>(sH, sAB, tid);
    __syncthreads();

    // ---- phase 2b: residual term 1: (hn @ w_out), wave-parallel (4 rows/wave)
    const int rw = wv << 2;
    float racc[4][DD];
    #pragma unroll
    for (int rr = 0; rr < 4; ++rr)
        #pragma unroll
        for (int d = 0; d < DD; ++d) racc[rr][d] = 0.f;
    #pragma unroll
    for (int rr = 0; rr < 4; ++rr)
        rowdot6(sH + (rw + rr) * SROW, w_out, lane, racc[rr]);

    // ---- phase 3: y = hn @ Kmat via v_wmma_f32_16x16x32_bf16
    // wave 'wv' owns 2 m-tiles x 8 n-tiles; each B frag feeds 2 WMMAs.
    v8f zero = {};
    v8f acc[16];
    #pragma unroll
    for (int t = 0; t < 16; ++t) acc[t] = zero;

    for (int kc = 0; kc < 32; ++kc) {
        // prefetch all 8 B fragments (16 global_load_b128 in flight)
        v16bf bf[8];
        #pragma unroll
        for (int t = 0; t < 8; ++t) {
            int nt = (wv << 3) + t;
            bf[t] = *(const v16bf*)(kswz +
                (((size_t)(kc * 64 + nt)) << 9) + (lane << 4));
        }
        #pragma unroll
        for (int mt = 0; mt < 2; ++mt) {
            // A fragment: conflict-free ds_load_b64 x4 from staging
            union { v16bf v; uint2 q[4]; } ua;
            const unsigned int* ak = sAB + (kc * 2 + mt) * AKC + lane * ASTR;
            #pragma unroll
            for (int j = 0; j < 4; ++j) ua.q[j] = *(const uint2*)(ak + 2 * j);
            #pragma unroll
            for (int t = 0; t < 8; ++t)
                acc[mt * 8 + t] = __builtin_amdgcn_wmma_f32_16x16x32_bf16(
                    false, ua.v, false, bf[t], (short)0, acc[mt * 8 + t], false, false);
        }
    }
    __syncthreads();   // all waves done reading hn from sH

    // ---- phase 4: stage y into sH (C/D layout: lanes 0-15 -> M=j, 16-31 -> M=j+8)
    {
        int rofs = (lane >> 4) << 3;
        #pragma unroll
        for (int mt = 0; mt < 2; ++mt) {
            #pragma unroll
            for (int t = 0; t < 8; ++t) {
                int col = (wv << 7) + (t << 4) + (lane & 15);
                #pragma unroll
                for (int j = 0; j < 8; ++j)
                    sH[(mt * 16 + rofs + j) * SROW + col] = acc[mt * 8 + t][j];
            }
        }
    }
    __syncthreads();

    // ---- phase 5: manifold-normalize y in place
    norm_tile<false>(sH, sAB, tid);
    __syncthreads();

    // ---- phase 6: residual term 2 + butterfly reduce + output
    #pragma unroll
    for (int rr = 0; rr < 4; ++rr)
        rowdot6(sH + (rw + rr) * SROW, w_out, lane, racc[rr]);

    #pragma unroll
    for (int rr = 0; rr < 4; ++rr) {
        #pragma unroll
        for (int d = 0; d < DD; ++d) {
            float v = racc[rr][d];
            #pragma unroll
            for (int m = 16; m >= 1; m >>= 1) v += __shfl_xor(v, m, 32);
            if (lane == rr * DD + d) {
                size_t oi = (size_t)(rowBase + rw + rr) * DD + d;
                out[oi] = x[oi] + v + b_out[d];
            }
        }
    }
}

extern "C" void kernel_launch(void* const* d_in, const int* in_sizes, int n_in,
                              void* d_out, int out_size, void* d_ws, size_t ws_size,
                              hipStream_t stream) {
    const float* x     = (const float*)d_in[0];   // [32,256,5,6]
    const float* w_in  = (const float*)d_in[1];   // [6,1024]
    const float* b_in  = (const float*)d_in[2];   // [1024]
    const float* w_out = (const float*)d_in[3];   // [1024,6]
    const float* b_out = (const float*)d_in[4];   // [6]
    const float* gaw   = (const float*)d_in[5];   // [32,32,32]
    float* out = (float*)d_out;
    __bf16* kswz = (__bf16*)d_ws;                 // 1024*1024 bf16 = 2 MB

    build_k_swz<<<(1024 * 1024) / 256, 256, 0, stream>>>(gaw, kswz);
    fused_ga<<<NROWS / MT, 256, 0, stream>>>(x, w_in, b_in, w_out, b_out, kswz, out);
}
